// SelfAttention_75574244540579
// MI455X (gfx1250) — matrix operand order repaired and checked by reading, not compiled, not run on hardware
//
#include <hip/hip_runtime.h>

typedef __bf16 bf16;
typedef __attribute__((ext_vector_type(16))) __bf16 bf16x16;
typedef __attribute__((ext_vector_type(8)))  __bf16 bf16x8;
typedef __attribute__((ext_vector_type(8)))  float  f32x8;

#define NTOK 8192
#define EDIM 1024

union Frag { bf16x16 v; bf16x8 h[2]; };

__device__ __forceinline__ f32x8 wmma_bf16(const Frag& a, const Frag& b, f32x8 c) {
  // D = A(16x32 bf16) x B(32x16 bf16) + C(16x16 f32)
  return __builtin_amdgcn_wmma_f32_16x16x32_bf16(false, a.v, false, b.v,
                                                 (short)0, c, false, false);
}

// Async direct global->LDS 16-byte copy (CDNA5, tracked by ASYNCcnt).
// INST_OFFSET is added to BOTH the LDS and the global address (ISA 08, 4.4).
template <int OFF>
__device__ __forceinline__ void async_g2l_16(unsigned lds_off, const bf16* g) {
  asm volatile("global_load_async_to_lds_b128 %0, %1, off offset:%2"
               :: "v"(lds_off), "v"(g), "i"(OFF) : "memory");
}
__device__ __forceinline__ void wait_async0() {
  asm volatile("s_wait_asynccnt 0x0" ::: "memory");
}

// ---------------------------------------------------------------------------
// fp32 -> bf16 elementwise convert
__global__ void cvt_f32_bf16(const float* __restrict__ src, bf16* __restrict__ dst, int n) {
  int i = blockIdx.x * blockDim.x + threadIdx.x;
  int stride = gridDim.x * blockDim.x;
  for (; i < n; i += stride) dst[i] = (bf16)src[i];
}

// W: [3][E][E] (k-major rows) -> Wt: [3][col][k] bf16 (so B-frag loads are contiguous in k)
__global__ void wcvt_transpose(const float* __restrict__ w, bf16* __restrict__ wt) {
  int idx = blockIdx.x * blockDim.x + threadIdx.x;
  if (idx >= 3 * EDIM * EDIM) return;
  int n   = idx / (EDIM * EDIM);
  int r   = idx % (EDIM * EDIM);
  int col = r / EDIM;
  int k   = r % EDIM;
  wt[idx] = (bf16)w[(size_t)n * EDIM * EDIM + (size_t)k * EDIM + col];
}

// ---------------------------------------------------------------------------
// gates[p][row][n] = softmax_n(x[row,:] . Wg_p[:,n]); one wave per row
__global__ __launch_bounds__(256) void gate_kernel(
    const float* __restrict__ x,
    const float* __restrict__ wgq, const float* __restrict__ wgk, const float* __restrict__ wgv,
    float* __restrict__ gates) {
  int lane = threadIdx.x & 31;
  int row  = blockIdx.x * 8 + (threadIdx.x >> 5);
  int p    = blockIdx.y;
  const float* wg = (p == 0) ? wgq : (p == 1) ? wgk : wgv;
  const float* xr = x + (size_t)row * EDIM;
  float a0 = 0.f, a1 = 0.f, a2 = 0.f;
  for (int j = 0; j < EDIM; j += 32) {
    float xv = xr[j + lane];
    const float* wr = wg + (size_t)(j + lane) * 3;
    a0 += xv * wr[0]; a1 += xv * wr[1]; a2 += xv * wr[2];
  }
  for (int off = 16; off; off >>= 1) {
    a0 += __shfl_xor(a0, off, 32);
    a1 += __shfl_xor(a1, off, 32);
    a2 += __shfl_xor(a2, off, 32);
  }
  if (lane == 0) {
    float mx = fmaxf(a0, fmaxf(a1, a2));
    float e0 = __expf(a0 - mx), e1 = __expf(a1 - mx), e2 = __expf(a2 - mx);
    float inv = 1.0f / (e0 + e1 + e2);
    float* g = gates + ((size_t)p * NTOK + row) * 3;
    g[0] = e0 * inv; g[1] = e1 * inv; g[2] = e2 * inv;
  }
}

// ---------------------------------------------------------------------------
// MoE GEMM: out[row,col] = sum_n gate[row,n] * (x[row,:] @ W_n[:,col])
// Gate folded into A: stage 3 gate-prescaled copies of the x tile.
// B tiles staged with async global->LDS (no VGPR round-trip).
// Block: 256 thr = 8 waves, 128x128 tile; wave = 32 rows x 64 cols (2x4 WMMA tiles).
__global__ __launch_bounds__(256) void moe_gemm_kernel(
    const bf16*  __restrict__ xb,     // [NTOK][EDIM]
    const bf16*  __restrict__ wt,     // [3][col][k]
    const float* __restrict__ gates,  // [NTOK][3]
    bf16* __restrict__ outp,          // [NTOK][EDIM] or transposed [EDIM][NTOK]
    int transpose_out) {
  __shared__ bf16 As[3][128][40];
  __shared__ bf16 Bs[3][128][40];

  const int tid  = threadIdx.x;
  const int lane = tid & 31;
  const int wid  = tid >> 5;
  const int rw   = wid >> 1;            // 0..3  (row wave)
  const int cw   = wid & 1;             // 0..1  (col wave)
  const int ml   = lane & 15;
  const int hi   = lane >> 4;           // 0/1
  const int row0 = blockIdx.y * 128;
  const int col0 = blockIdx.x * 128;

  const int srow  = tid >> 1;           // staging row/col 0..127
  const int khalf = (tid & 1) * 16;     // k half 0 / 16

  float gg[3];
  gg[0] = gates[(size_t)(row0 + srow) * 3 + 0];
  gg[1] = gates[(size_t)(row0 + srow) * 3 + 1];
  gg[2] = gates[(size_t)(row0 + srow) * 3 + 2];

  f32x8 acc[2][4];
  for (int i = 0; i < 2; ++i)
    for (int j = 0; j < 4; ++j)
      acc[i][j] = f32x8{0.f, 0.f, 0.f, 0.f, 0.f, 0.f, 0.f, 0.f};

  for (int kb = 0; kb < EDIM; kb += 32) {
    // ---- stage B via async global->LDS (overlaps with A's VALU work) ----
    for (int n = 0; n < 3; ++n) {
      const bf16* wp = wt + ((size_t)n * EDIM + (col0 + srow)) * EDIM + kb + khalf;
      unsigned ldsb = (unsigned)(size_t)&Bs[n][srow][khalf];
      async_g2l_16<0>(ldsb, wp);
      async_g2l_16<16>(ldsb, wp);
    }
    // ---- stage A (gate-prescaled x) through VGPRs ----
    const bf16* xp = &xb[(size_t)(row0 + srow) * EDIM + kb + khalf];
    bf16x8 xlo = *(const bf16x8*)&xp[0];
    bf16x8 xhi = *(const bf16x8*)&xp[8];
    if (kb + 32 < EDIM) __builtin_prefetch(xp + 32, 0, 1);
    for (int n = 0; n < 3; ++n) {
      bf16x8 slo, shi;
      for (int t = 0; t < 8; ++t) {
        slo[t] = (bf16)((float)xlo[t] * gg[n]);
        shi[t] = (bf16)((float)xhi[t] * gg[n]);
      }
      *(bf16x8*)&As[n][srow][khalf]     = slo;
      *(bf16x8*)&As[n][srow][khalf + 8] = shi;
    }
    wait_async0();       // our async B transfers have landed in LDS
    __syncthreads();     // everyone's A stores + B asyncs visible

    for (int n = 0; n < 3; ++n) {
      Frag af[2];
      for (int i = 0; i < 2; ++i) {
        const bf16* ap = &As[n][rw * 32 + i * 16 + ml][hi * 8];
        af[i].h[0] = *(const bf16x8*)ap;          // K 0-7 / 8-15
        af[i].h[1] = *(const bf16x8*)(ap + 16);   // K 16-23 / 24-31
      }
      Frag bfr[4];
      for (int j = 0; j < 4; ++j) {
        const bf16* bp = &Bs[n][cw * 64 + j * 16 + ml][hi * 8];
        bfr[j].h[0] = *(const bf16x8*)bp;         // K 0-7 / 8-15
        bfr[j].h[1] = *(const bf16x8*)(bp + 16);  // K 16-23 / 24-31
      }
      for (int i = 0; i < 2; ++i)
        for (int j = 0; j < 4; ++j)
          acc[i][j] = wmma_bf16(af[i], bfr[j], acc[i][j]);
    }
    __syncthreads();
  }

  // epilogue: C layout -> (M = v + 8*hi, N = ml)
  for (int i = 0; i < 2; ++i)
    for (int j = 0; j < 4; ++j) {
      int Cg = col0 + cw * 64 + j * 16 + ml;
      int Rb = row0 + rw * 32 + i * 16 + hi * 8;
      for (int v = 0; v < 8; ++v) {
        bf16 val = (bf16)acc[i][j][v];
        if (!transpose_out) outp[(size_t)(Rb + v) * EDIM + Cg] = val;
        else                outp[(size_t)Cg * NTOK + (Rb + v)] = val;
      }
    }
}

// ---------------------------------------------------------------------------
// Flash attention: block = 8 waves x 16 q-rows = 128 q rows; D = 64; keys in 64-blocks.
__global__ __launch_bounds__(256) void attn_kernel(
    const bf16* __restrict__ qb,   // [NTOK][EDIM]
    const bf16* __restrict__ kbuf, // [NTOK][EDIM]
    const bf16* __restrict__ vt,   // [EDIM][NTOK]  (transposed V)
    float* __restrict__ outp) {    // [NTOK][EDIM]
  __shared__ bf16 P[8][16][72];

  const int tid  = threadIdx.x;
  const int lane = tid & 31;
  const int w    = tid >> 5;
  const int ml   = lane & 15;
  const int hi   = lane >> 4;
  const int b    = blockIdx.y;
  const int mi   = b >> 4;       // batch
  const int h    = b & 15;       // head
  const int tok0 = mi * 2048;
  const int q0   = blockIdx.x * 128 + w * 16;
  const int hc   = h * 64;

  // Q fragments (kept in registers for all key blocks)
  Frag qf[2];
  const bf16* qrow = qb + (size_t)(tok0 + q0 + ml) * EDIM + hc;
  for (int ka = 0; ka < 2; ++ka) {
    int dlo = ka * 32 + hi * 8;
    qf[ka].h[0] = *(const bf16x8*)&qrow[dlo];
    qf[ka].h[1] = *(const bf16x8*)&qrow[dlo + 16];
  }

  f32x8 o[4];
  for (int nt = 0; nt < 4; ++nt)
    o[nt] = f32x8{0.f, 0.f, 0.f, 0.f, 0.f, 0.f, 0.f, 0.f};
  float mrun[8], lrun[8];
  for (int j = 0; j < 8; ++j) { mrun[j] = -1e30f; lrun[j] = 0.f; }

  for (int kb = 0; kb < 2048; kb += 64) {
    // S = (Q K^T) / 8
    f32x8 s[4];
    for (int nt = 0; nt < 4; ++nt) {
      s[nt] = f32x8{0.f, 0.f, 0.f, 0.f, 0.f, 0.f, 0.f, 0.f};
      for (int ka = 0; ka < 2; ++ka) {
        Frag kf;
        const bf16* krow = kbuf + (size_t)(tok0 + kb + nt * 16 + ml) * EDIM
                         + hc + ka * 32 + hi * 8;
        kf.h[0] = *(const bf16x8*)&krow[0];
        kf.h[1] = *(const bf16x8*)&krow[16];
        s[nt] = wmma_bf16(qf[ka], kf, s[nt]);
      }
      s[nt] *= 0.125f;  // 1/sqrt(64)
    }
    // online softmax (rows live in 16-lane halves)
    float rmax[8], alpha[8];
    for (int j = 0; j < 8; ++j) {
      float m0 = fmaxf(fmaxf(s[0][j], s[1][j]), fmaxf(s[2][j], s[3][j]));
      for (int off = 1; off < 16; off <<= 1)
        m0 = fmaxf(m0, __shfl_xor(m0, off, 32));
      rmax[j] = m0;
      float nm = fmaxf(mrun[j], m0);
      alpha[j] = __expf(mrun[j] - nm);
      mrun[j] = nm;
    }
    for (int nt = 0; nt < 4; ++nt)
      for (int j = 0; j < 8; ++j)
        s[nt][j] = __expf(s[nt][j] - mrun[j]);
    for (int j = 0; j < 8; ++j) {
      float t = s[0][j] + s[1][j] + s[2][j] + s[3][j];
      for (int off = 1; off < 16; off <<= 1)
        t += __shfl_xor(t, off, 32);
      lrun[j] = lrun[j] * alpha[j] + t;
    }
    for (int nt = 0; nt < 4; ++nt)
      for (int j = 0; j < 8; ++j)
        o[nt][j] *= alpha[j];

    // repack P from C layout to A layout through per-wave LDS
    for (int nt = 0; nt < 4; ++nt)
      for (int j = 0; j < 8; ++j)
        P[w][j + hi * 8][nt * 16 + ml] = (bf16)s[nt][j];
    __builtin_amdgcn_wave_barrier();
    Frag pf[2];
    for (int ka = 0; ka < 2; ++ka) {
      int klo = ka * 32 + hi * 8;
      pf[ka].h[0] = *(const bf16x8*)&P[w][ml][klo];
      pf[ka].h[1] = *(const bf16x8*)&P[w][ml][klo + 16];
    }
    // O += P V
    for (int nt = 0; nt < 4; ++nt)
      for (int ka = 0; ka < 2; ++ka) {
        Frag vf;
        const bf16* vrow = vt + (size_t)(hc + nt * 16 + ml) * NTOK
                         + tok0 + kb + ka * 32 + hi * 8;
        vf.h[0] = *(const bf16x8*)&vrow[0];
        vf.h[1] = *(const bf16x8*)&vrow[16];
        o[nt] = wmma_bf16(pf[ka], vf, o[nt]);
      }
  }

  for (int nt = 0; nt < 4; ++nt)
    for (int j = 0; j < 8; ++j) {
      int row = tok0 + q0 + j + hi * 8;
      outp[(size_t)row * EDIM + hc + nt * 16 + ml] = o[nt][j] / lrun[j];
    }
}

// ---------------------------------------------------------------------------
extern "C" void kernel_launch(void* const* d_in, const int* in_sizes, int n_in,
                              void* d_out, int out_size, void* d_ws, size_t ws_size,
                              hipStream_t stream) {
  const float* x   = (const float*)d_in[0];
  const float* Wq  = (const float*)d_in[1];
  const float* Wgq = (const float*)d_in[3];
  const float* Wk  = (const float*)d_in[5];
  const float* Wgk = (const float*)d_in[7];
  const float* Wv  = (const float*)d_in[9];
  const float* Wgv = (const float*)d_in[11];
  float* out = (float*)d_out;

  char* ws = (char*)d_ws;
  size_t off = 0;
  bf16* xbb = (bf16*)(ws + off); off += (size_t)NTOK * EDIM * 2;           // 16 MB
  bf16* wtq = (bf16*)(ws + off); off += (size_t)3 * EDIM * EDIM * 2;       // 6 MB
  bf16* wtk = (bf16*)(ws + off); off += (size_t)3 * EDIM * EDIM * 2;
  bf16* wtv = (bf16*)(ws + off); off += (size_t)3 * EDIM * EDIM * 2;
  float* gates = (float*)(ws + off); off += (size_t)3 * NTOK * 3 * 4;
  bf16* qb = (bf16*)(ws + off); off += (size_t)NTOK * EDIM * 2;
  bf16* kb = (bf16*)(ws + off); off += (size_t)NTOK * EDIM * 2;
  bf16* vt = (bf16*)(ws + off); off += (size_t)NTOK * EDIM * 2;

  // 1. convert inputs to bf16 (weights transposed to col-major)
  cvt_f32_bf16<<<512, 256, 0, stream>>>(x, xbb, NTOK * EDIM);
  int wn = 3 * EDIM * EDIM;
  wcvt_transpose<<<(wn + 255) / 256, 256, 0, stream>>>(Wq, wtq);
  wcvt_transpose<<<(wn + 255) / 256, 256, 0, stream>>>(Wk, wtk);
  wcvt_transpose<<<(wn + 255) / 256, 256, 0, stream>>>(Wv, wtv);

  // 2. gates (fp32 path, tiny)
  gate_kernel<<<dim3(NTOK / 8, 3), 256, 0, stream>>>(x, Wgq, Wgk, Wgv, gates);

  // 3. MoE GEMMs (V stored transposed for attention B-fragments)
  dim3 gg(EDIM / 128, NTOK / 128);
  moe_gemm_kernel<<<gg, 256, 0, stream>>>(xbb, wtq, gates + 0 * NTOK * 3, qb, 0);
  moe_gemm_kernel<<<gg, 256, 0, stream>>>(xbb, wtk, gates + 1 * NTOK * 3, kb, 0);
  moe_gemm_kernel<<<gg, 256, 0, stream>>>(xbb, wtv, gates + 2 * NTOK * 3, vt, 1);

  // 4. flash attention, fp32 out
  attn_kernel<<<dim3(2048 / 128, 4 * 16), 256, 0, stream>>>(qb, kb, vt, out);
}
